// LinearAttention_11879879541013
// MI455X (gfx1250) — matrix operand order repaired and checked by reading, not compiled, not run on hardware
//
#include <hip/hip_runtime.h>
#include <hip/hip_bf16.h>

// ---------------------------------------------------------------------------
// Linear attention for MI455X (gfx1250), wave32 + v_wmma_f32_16x16x32_f16.
// Compute-bound on QKV projection (309 GFLOP of 343); f16 WMMA w/ f32 accum.
// A panels staged once in LDS (f16) per K-chunk; kv accumulated via 8x8
// register tiles fed by ds_load_b128.
// ---------------------------------------------------------------------------

typedef __attribute__((ext_vector_type(16))) _Float16 v16h;
typedef __attribute__((ext_vector_type(8)))  _Float16 v8h;
typedef __attribute__((ext_vector_type(4)))  _Float16 v4h;
typedef __attribute__((ext_vector_type(8)))  float    v8f;
typedef __attribute__((ext_vector_type(4)))  float    v4f;

constexpr int L_SEQ  = 8192;
constexpr int DMODEL = 512;
constexpr int NH     = 8;
constexpr int DHEAD  = 64;
constexpr int K3     = 1536;   // 3*DMODEL
constexpr int ROWS   = 32;     // rows per workgroup chunk

#define WMMA(a, b, c) \
  __builtin_amdgcn_wmma_f32_16x16x32_f16(false, (a), false, (b), (short)0, (c), false, false)

// ---- Fragment loaders (contiguous k = 16*(lane/16) + s, same order for A and B) -----------------

// A (16x32 f16, MxK): lane&15 = M row; lanes 0-15 hold K {k0..k0+15},
// lanes 16-31 hold K {k0+16..31}. Source: f16 row-major LDS tile.
static __device__ inline v16h load_a_f16(const _Float16* base, int ld, int row0,
                                         int k0, int lane) {
  const int m  = lane & 15;
  const int kk = k0 + ((lane >> 4) << 4);
  const _Float16* p = base + (row0 + m) * ld + kk;
  v16h a;
#pragma unroll
  for (int i = 0; i < 8; ++i) {
    a[i]     = p[i];
    a[8 + i] = p[8 + i];
  }
  return a;
}

// B (32x16 f16, KxN): rows striped across lanes; lane = (K>=16)*16 + N, and
// each lane holds a contiguous 16-long K segment of column N, i.e. a
// contiguous row of B^T. Source w is B^T row-major: w[n][k], leading dim ldk.
static __device__ inline v16h load_bT(const _Float16* w, int ldk, int n0,
                                      int k0, int lane) {
  const int n  = n0 + (lane & 15);
  const int kk = k0 + ((lane >> 4) << 4);
  const _Float16* p = w + (size_t)n * (size_t)ldk + kk;
  v16h b;
#pragma unroll
  for (int i = 0; i < 16; ++i) b[i] = p[i];
  return b;
}

static __device__ inline float elu1(float x) {
  return x > 0.f ? x + 1.f : __expf(x);
}

// Cooperatively stage a ROWS x 512 f32 panel into LDS as f16 (row-major).
static __device__ inline void stage_panel(const float* __restrict__ src,
                                          int row0, _Float16* panel) {
  for (int idx = threadIdx.x * 4; idx < ROWS * DMODEL; idx += 256 * 4) {
    const int s = idx >> 9;
    const int c = idx & (DMODEL - 1);
    v4f val = *(const v4f*)(src + (size_t)(row0 + s) * DMODEL + c);
    v4h hv;
#pragma unroll
    for (int j = 0; j < 4; ++j) hv[j] = (_Float16)val[j];
    *(v4h*)(panel + idx) = hv;
  }
}

// ---- Small utility kernels --------------------------------------------------

__global__ void zero_f32_kernel(float* p, int n) {
  int i = blockIdx.x * blockDim.x + threadIdx.x;
  if (i < n) p[i] = 0.f;
}

__global__ void cvt_f16_kernel(const float* src, _Float16* dst, int n) {
  int i = blockIdx.x * blockDim.x + threadIdx.x;
  if (i < n) dst[i] = (_Float16)src[i];
}

// ---- Pass 1: k/v projection + kv[b,h,m,d] & k_sum accumulation --------------
// 8 waves; wave w covers 128 output cols (global qkv cols 512 + w*128).

__global__ void __launch_bounds__(256) pass1_kernel(
    const float* __restrict__ q, const float* __restrict__ k,
    const float* __restrict__ v, const _Float16* __restrict__ wqkv16,
    float* __restrict__ kv_f, float* __restrict__ ksum) {
  // First 32KB: staged A panel during GEMM. Afterwards (aliased, barriered):
  // full 64KB = k_feat/v staging [s][0..511]=k_feat, [512..1023]=v.
  __shared__ __align__(16) _Float16 smem[ROWS * 1024];
  _Float16* panel = smem;
  _Float16* kvl   = smem;

  const int lane = threadIdx.x & 31;
  const int w    = threadIdx.x >> 5;
  const int row0 = blockIdx.x * ROWS;
  const int b    = row0 / L_SEQ;

  v8f acc[2][8];
#pragma unroll
  for (int mf = 0; mf < 2; ++mf)
#pragma unroll
    for (int nf = 0; nf < 8; ++nf)
#pragma unroll
      for (int i = 0; i < 8; ++i) acc[mf][nf][i] = 0.f;

  for (int kc = 0; kc < K3; kc += DMODEL) {
    const float* src = (kc == 0) ? q : ((kc == DMODEL) ? k : v);
    __syncthreads();  // prior chunk consumers done
    stage_panel(src, row0, panel);
    __syncthreads();
    for (int kk = 0; kk < DMODEL; kk += 32) {
      v16h a0 = load_a_f16(panel, DMODEL, 0, kk, lane);
      v16h a1 = load_a_f16(panel, DMODEL, 16, kk, lane);
      const int k0 = kc + kk;
#pragma unroll
      for (int nf = 0; nf < 8; ++nf) {
        v16h bf = load_bT(wqkv16, K3, DMODEL + w * 128 + nf * 16, k0, lane);
        acc[0][nf] = WMMA(a0, bf, acc[0][nf]);
        acc[1][nf] = WMMA(a1, bf, acc[1][nf]);
      }
    }
  }
  __syncthreads();  // all panel reads done; smem now reused as kvl

  const bool is_k = (w < 4);  // cols [0,512) of the 1024 are the k projection
#pragma unroll
  for (int mf = 0; mf < 2; ++mf)
#pragma unroll
    for (int nf = 0; nf < 8; ++nf) {
      const int col = w * 128 + nf * 16 + (lane & 15);
#pragma unroll
      for (int r = 0; r < 8; ++r) {
        const int s = 16 * mf + r + ((lane >> 4) << 3);
        float val = acc[mf][nf][r];
        if (is_k) val = elu1(val);
        kvl[s * 1024 + col] = (_Float16)val;
      }
    }
  __syncthreads();

  // kv[b][h][m][d] += sum_s k_feat[s][h*64+d] * v[s][h*64+m]
  // 8x8 register tiles; per s-step: 2x ds_load_b128 feeds 64 FMAs.
  float* kvb = kv_f + (size_t)b * (NH * DHEAD * DHEAD);
  for (int tile = threadIdx.x; tile < 512; tile += 256) {
    const int h  = tile >> 6;
    const int d0 = ((tile >> 3) & 7) * 8;
    const int m0 = (tile & 7) * 8;
    float accv[8][8];
#pragma unroll
    for (int mi = 0; mi < 8; ++mi)
#pragma unroll
      for (int di = 0; di < 8; ++di) accv[mi][di] = 0.f;
    for (int s = 0; s < ROWS; ++s) {
      v8h kf = *(const v8h*)&kvl[s * 1024 + h * 64 + d0];
      v8h vf = *(const v8h*)&kvl[s * 1024 + 512 + h * 64 + m0];
      float kfl[8], vfl[8];
#pragma unroll
      for (int i = 0; i < 8; ++i) {
        kfl[i] = (float)kf[i];
        vfl[i] = (float)vf[i];
      }
#pragma unroll
      for (int mi = 0; mi < 8; ++mi)
#pragma unroll
        for (int di = 0; di < 8; ++di) accv[mi][di] += vfl[mi] * kfl[di];
    }
    float* dst = kvb + ((h * 64 + m0) * 64 + d0);  // kv[h][m][d]
#pragma unroll
    for (int mi = 0; mi < 8; ++mi)
#pragma unroll
      for (int di = 0; di < 8; ++di)
        atomicAdd(&dst[mi * 64 + di], accv[mi][di]);
  }

  // k_sum[b][c] += sum_s k_feat[s][c]
  for (int c = threadIdx.x * 2; c < DMODEL; c += 512) {
    float s0 = 0.f, s1 = 0.f;
#pragma unroll 4
    for (int s = 0; s < ROWS; ++s) {
      s0 += (float)kvl[s * 1024 + c];
      s1 += (float)kvl[s * 1024 + c + 1];
    }
    atomicAdd(&ksum[b * DMODEL + c], s0);
    atomicAdd(&ksum[b * DMODEL + c + 1], s1);
  }
}

// ---- Pass 2 (fused): q proj + elu1 -> z -> attn = z*(q@kv^T) -> out proj ----
// 8 waves; wave w covers 64 cols == exactly head w for the attn phase.

__global__ void __launch_bounds__(256) pass2_kernel(
    const float* __restrict__ q, const float* __restrict__ k,
    const float* __restrict__ v, const _Float16* __restrict__ wqkv16,
    const _Float16* __restrict__ wout16, const _Float16* __restrict__ kv16,
    const float* __restrict__ ksum, const float* __restrict__ bout,
    float* __restrict__ out) {
  // 32KB buffer: staged A panel during phase A GEMM; afterwards q_feat, then
  // attn (all aliased with barriers between role changes).
  __shared__ __align__(16) _Float16 smem[ROWS * DMODEL];
  __shared__ float zl[ROWS * NH];
  _Float16* panel = smem;
  _Float16* qa    = smem;

  const int lane = threadIdx.x & 31;
  const int w    = threadIdx.x >> 5;
  const int row0 = blockIdx.x * ROWS;
  const int b    = row0 / L_SEQ;

  // Phase A: q_feat = elu1(x @ w_qkv[0:512].T)
  v8f acc[2][4];
#pragma unroll
  for (int mf = 0; mf < 2; ++mf)
#pragma unroll
    for (int nf = 0; nf < 4; ++nf)
#pragma unroll
      for (int i = 0; i < 8; ++i) acc[mf][nf][i] = 0.f;

  for (int kc = 0; kc < K3; kc += DMODEL) {
    const float* src = (kc == 0) ? q : ((kc == DMODEL) ? k : v);
    __syncthreads();
    stage_panel(src, row0, panel);
    __syncthreads();
    for (int kk = 0; kk < DMODEL; kk += 32) {
      v16h a0 = load_a_f16(panel, DMODEL, 0, kk, lane);
      v16h a1 = load_a_f16(panel, DMODEL, 16, kk, lane);
      const int k0 = kc + kk;
#pragma unroll
      for (int nf = 0; nf < 4; ++nf) {
        v16h bf = load_bT(wqkv16, K3, w * 64 + nf * 16, k0, lane);
        acc[0][nf] = WMMA(a0, bf, acc[0][nf]);
        acc[1][nf] = WMMA(a1, bf, acc[1][nf]);
      }
    }
  }
  __syncthreads();  // panel reads done; smem becomes q_feat
#pragma unroll
  for (int mf = 0; mf < 2; ++mf)
#pragma unroll
    for (int nf = 0; nf < 4; ++nf) {
      const int col = w * 64 + nf * 16 + (lane & 15);
#pragma unroll
      for (int r = 0; r < 8; ++r) {
        const int s = 16 * mf + r + ((lane >> 4) << 3);
        qa[s * DMODEL + col] = (_Float16)elu1(acc[mf][nf][r]);
      }
    }
  __syncthreads();

  // Phase B: z[r][h] = 1 / (q_feat[r,h,:] . k_sum[b,h,:] + eps)
  {
    const int r = threadIdx.x >> 3;
    const int h = threadIdx.x & 7;
    const float* ks = ksum + b * DMODEL + h * 64;
    float sum = 0.f;
#pragma unroll
    for (int d8 = 0; d8 < DHEAD; d8 += 8) {
      v8h qv = *(const v8h*)&qa[r * DMODEL + h * 64 + d8];
#pragma unroll
      for (int j = 0; j < 8; ++j) sum += (float)qv[j] * ks[d8 + j];
    }
    zl[r * NH + h] = 1.f / (sum + 1e-6f);
  }
  __syncthreads();

  // Phase C: attn[r][h*64+m] = z * sum_d q_feat[r][h*64+d] * kv[b,h,m,d]
  v8f acc2[2][4];
#pragma unroll
  for (int mf = 0; mf < 2; ++mf)
#pragma unroll
    for (int nf = 0; nf < 4; ++nf)
#pragma unroll
      for (int i = 0; i < 8; ++i) acc2[mf][nf][i] = 0.f;

  const _Float16* kvh = kv16 + (size_t)(b * NH + w) * (DHEAD * DHEAD); // [m][d]
#pragma unroll
  for (int kstep = 0; kstep < 2; ++kstep) {
    v16h a0 = load_a_f16(qa, DMODEL, 0, w * 64 + kstep * 32, lane);
    v16h a1 = load_a_f16(qa, DMODEL, 16, w * 64 + kstep * 32, lane);
#pragma unroll
    for (int nf = 0; nf < 4; ++nf) {
      // B[k=d][n=m] = kv[m][d]; kvh is exactly B^T row-major, ld = 64
      v16h bf = load_bT(kvh, DHEAD, nf * 16, kstep * 32, lane);
      acc2[0][nf] = WMMA(a0, bf, acc2[0][nf]);
      acc2[1][nf] = WMMA(a1, bf, acc2[1][nf]);
    }
  }
  __syncthreads();  // all q_feat reads done before attn overwrites qa
#pragma unroll
  for (int mf = 0; mf < 2; ++mf)
#pragma unroll
    for (int nf = 0; nf < 4; ++nf) {
      const int col = w * 64 + nf * 16 + (lane & 15);
#pragma unroll
      for (int r = 0; r < 8; ++r) {
        const int s = 16 * mf + r + ((lane >> 4) << 3);
        qa[s * DMODEL + col] = (_Float16)(acc2[mf][nf][r] * zl[s * NH + w]);
      }
    }
  __syncthreads();

  // Phase D: out = attn @ w_out.T + b_out
  v8f acc3[2][4];
#pragma unroll
  for (int mf = 0; mf < 2; ++mf)
#pragma unroll
    for (int nf = 0; nf < 4; ++nf)
#pragma unroll
      for (int i = 0; i < 8; ++i) acc3[mf][nf][i] = 0.f;

  for (int k0 = 0; k0 < DMODEL; k0 += 32) {
    v16h a0 = load_a_f16(qa, DMODEL, 0, k0, lane);
    v16h a1 = load_a_f16(qa, DMODEL, 16, k0, lane);
#pragma unroll
    for (int nf = 0; nf < 4; ++nf) {
      v16h bf = load_bT(wout16, DMODEL, w * 64 + nf * 16, k0, lane);
      acc3[0][nf] = WMMA(a0, bf, acc3[0][nf]);
      acc3[1][nf] = WMMA(a1, bf, acc3[1][nf]);
    }
  }
#pragma unroll
  for (int mf = 0; mf < 2; ++mf)
#pragma unroll
    for (int nf = 0; nf < 4; ++nf) {
      const int col = w * 64 + nf * 16 + (lane & 15);
#pragma unroll
      for (int r = 0; r < 8; ++r) {
        const int s = 16 * mf + r + ((lane >> 4) << 3);
        out[(size_t)(row0 + s) * DMODEL + col] = acc3[mf][nf][r] + bout[col];
      }
    }
}

// ---------------------------------------------------------------------------

extern "C" void kernel_launch(void* const* d_in, const int* in_sizes, int n_in,
                              void* d_out, int out_size, void* d_ws,
                              size_t ws_size, hipStream_t stream) {
  const float* q    = (const float*)d_in[0];
  const float* k    = (const float*)d_in[1];
  const float* v    = (const float*)d_in[2];
  const float* wqkv = (const float*)d_in[3];
  const float* wout = (const float*)d_in[4];
  const float* bout = (const float*)d_in[5];
  float* out = (float*)d_out;

  // Workspace layout (~6.8 MB total; everything fits comfortably in L2)
  constexpr int NKV   = 8 * NH * DHEAD * DHEAD;  // 262144
  constexpr int NKSUM = 8 * DMODEL;              // 4096
  float*    kv_f   = (float*)d_ws;
  float*    ksum   = kv_f + NKV;
  _Float16* kv16   = (_Float16*)(ksum + NKSUM);
  _Float16* wqkv16 = kv16 + NKV;
  _Float16* wout16 = wqkv16 + K3 * K3;

  const int nz = NKV + NKSUM;  // kv_f and ksum are contiguous
  zero_f32_kernel<<<(nz + 255) / 256, 256, 0, stream>>>(kv_f, nz);
  cvt_f16_kernel<<<(K3 * K3 + 255) / 256, 256, 0, stream>>>(wqkv, wqkv16,
                                                            K3 * K3);
  cvt_f16_kernel<<<(DMODEL * DMODEL + 255) / 256, 256, 0, stream>>>(
      wout, wout16, DMODEL * DMODEL);

  const int nblocks = (8 * L_SEQ) / ROWS;  // 2048
  pass1_kernel<<<nblocks, 256, 0, stream>>>(q, k, v, wqkv16, kv_f, ksum);
  cvt_f16_kernel<<<(NKV + 255) / 256, 256, 0, stream>>>(kv_f, kv16, NKV);
  pass2_kernel<<<nblocks, 256, 0, stream>>>(q, k, v, wqkv16, wout16, kv16,
                                            ksum, bout, out);
}
